// CrossAtt0228_55439437857360
// MI455X (gfx1250) — matrix operand order repaired and checked by reading, not compiled, not run on hardware
//
#include <hip/hip_runtime.h>

#define BB 2
#define CIN 256
#define CI 128
#define NHEAD 4
#define HWSZ 2304
#define EPSV 1e-5f

typedef __attribute__((ext_vector_type(16))) _Float16 v16h;
typedef __attribute__((ext_vector_type(8)))  _Float16 v8h;
typedef __attribute__((ext_vector_type(8)))  float    v8f;
typedef __attribute__((ext_vector_type(2)))  float    v2f;

union F16F { v16h v; v8h h[2]; };

#define USE_ASYNC_LDS 1

__device__ __forceinline__ v8f wmma_f16(v16h a, v16h b, v8f c) {
  return __builtin_amdgcn_wmma_f32_16x16x32_f16(false, a, false, b, (short)0, c, false, false);
}

#if __has_builtin(__builtin_amdgcn_wmma_f32_16x16x4_f32)
#define HAVE_WMMA_F32X4 1
__device__ __forceinline__ v8f wmma_f32x4(v2f a, v2f b, v8f c) {
  return __builtin_amdgcn_wmma_f32_16x16x4_f32(false, a, false, b, (short)0, c, false, false);
}
#endif

#if USE_ASYNC_LDS
// CDNA5 async global->LDS copy (ASYNCcnt-tracked), 16B per lane.
__device__ __forceinline__ void async_b128(unsigned lds_off, const void* gaddr) {
  asm volatile("global_load_async_to_lds_b128 %0, %1, off"
               :: "v"(lds_off), "v"(gaddr) : "memory");
}
template <int N>
__device__ __forceinline__ void wait_async_n() {
#if __has_builtin(__builtin_amdgcn_s_wait_asynccnt)
  __builtin_amdgcn_s_wait_asynccnt(N);
#else
  asm volatile("s_wait_asynccnt %0" :: "i"(N) : "memory");
#endif
}
#endif

// ---------------------------------------------------------------------------
__global__ void bn_prep(const float* __restrict__ g, const float* __restrict__ be,
                        const float* __restrict__ m, const float* __restrict__ v,
                        const float* __restrict__ bias,
                        float* __restrict__ scale, float* __restrict__ shift, int n) {
  int i = blockIdx.x * blockDim.x + threadIdx.x;
  if (i < n) {
    float inv = g[i] * rsqrtf(v[i] + EPSV);
    scale[i] = inv;
    float sh = be[i] - m[i] * inv;
    if (bias) sh += bias[i] * inv;
    shift[i] = sh;
  }
}

__global__ void cvt_f16(const float* __restrict__ in, _Float16* __restrict__ out, int n) {
  int i = blockIdx.x * blockDim.x + threadIdx.x;
  if (i < n) out[i] = (_Float16)in[i];
}

// in: [K][N] f32 -> out: [N][K] f16 (32x32 LDS tile transpose)
__global__ __launch_bounds__(256) void transpose_cvt(
    const float* __restrict__ in, _Float16* __restrict__ out,
    int K, int N, long long ibs, long long obs) {
  __shared__ float tile[32][33];
  const int b = blockIdx.z;
  const int nb = blockIdx.x * 32, kb = blockIdx.y * 32;
  const float* ib = in + (long long)b * ibs;
  _Float16* ob = out + (long long)b * obs;
  #pragma unroll
  for (int j = 0; j < 4; ++j) {
    int kk = (threadIdx.x >> 5) + j * 8;
    int nn = threadIdx.x & 31;
    tile[kk][nn] = ib[(long long)(kb + kk) * N + nb + nn];
  }
  __syncthreads();
  #pragma unroll
  for (int j = 0; j < 4; ++j) {
    int nn = (threadIdx.x >> 5) + j * 8;
    int kk = threadIdx.x & 31;
    ob[(long long)(nb + nn) * K + kb + kk] = (_Float16)tile[kk][nn];
  }
}

// ---------------------------------------------------------------------------
// GEMM: W16[M,K] (f16, row-major) @ XT[N,K] (f16, transposed activations).
// Double-buffered LDS, async copies pipelined on ASYNCcnt (copy of tile t+1
// overlaps WMMA of tile t; in-order async completion => wait<=2 drains tile t).
// Outputs (each optional): Outf f32 [M,N]; Out16T f16 [N,M] (chain); Out16RM f16 [M,N].
// mode 0: +bias ; mode 1: *scale+shift ; mode 2: *scale+shift, relu.
__global__ __launch_bounds__(256) void gemm_wf16(
    const _Float16* __restrict__ W16, const _Float16* __restrict__ XT,
    float* __restrict__ Outf, _Float16* __restrict__ Out16T,
    _Float16* __restrict__ Out16RM,
    const float* __restrict__ p0, const float* __restrict__ p1,
    int M, int K, int N, long long bxs, long long bof, long long bo16, int mode) {
  __shared__ __attribute__((aligned(16))) _Float16 As[2][64 * 40];
  __shared__ __attribute__((aligned(16))) _Float16 Bs[2][64 * 40];
  const int b = blockIdx.z;
  const _Float16* XTb = XT + (long long)b * bxs;
  const int m0 = blockIdx.y * 64;
  const int n0 = blockIdx.x * 64;
  const int tid = threadIdx.x;
  const int lane = tid & 31, wave = tid >> 5;
  const int wm = wave & 3, wn = wave >> 2;
  const int half = lane >> 4, l16 = lane & 15;
  const int rr = tid >> 2, kc = (tid & 3) * 8;   // staging: 64 rows x 4 chunks of 8 halves
  const bool fullA = (m0 + 64 <= M);
  const bool avalid = (m0 + rr < M);

  auto stage = [&](int k0, _Float16* Ad, _Float16* Bd) {
    // A tile [64][32]
    if (fullA) {
#if USE_ASYNC_LDS
      async_b128((unsigned)(size_t)&Ad[rr * 40 + kc],
                 &W16[(long long)(m0 + rr) * K + k0 + kc]);
#else
      *(v8h*)&Ad[rr * 40 + kc] = *(const v8h*)&W16[(long long)(m0 + rr) * K + k0 + kc];
#endif
    } else {
      v8h av = {};
      if (avalid) av = *(const v8h*)&W16[(long long)(m0 + rr) * K + k0 + kc];
      *(v8h*)&Ad[rr * 40 + kc] = av;
    }
    // B tile [64 n][32 k] (contiguous rows of transposed activations)
#if USE_ASYNC_LDS
    async_b128((unsigned)(size_t)&Bd[rr * 40 + kc],
               &XTb[(long long)(n0 + rr) * K + k0 + kc]);
#else
    *(v8h*)&Bd[rr * 40 + kc] = *(const v8h*)&XTb[(long long)(n0 + rr) * K + k0 + kc];
#endif
  };

  v8f acc0 = {}, acc1 = {};
  const int nt = K >> 5;
  stage(0, As[0], Bs[0]);
  for (int t = 0; t < nt; ++t) {
    const _Float16* Ac = As[t & 1];
    const _Float16* Bc = Bs[t & 1];
    __syncthreads();  // everyone done reading buffer (t+1)&1 from iteration t-1
    if (t + 1 < nt) {
      stage((t + 1) << 5, As[(t + 1) & 1], Bs[(t + 1) & 1]);
#if USE_ASYNC_LDS
      if (fullA) wait_async_n<2>(); else wait_async_n<1>();
#endif
    } else {
#if USE_ASYNC_LDS
      wait_async_n<0>();
#endif
    }
    __syncthreads();  // all waves' copies of tile t have landed
    // fragments (contiguous k = 16*half + s in both A and B) + WMMA
    F16F a;
    a.h[0] = *(const v8h*)&Ac[(wm * 16 + l16) * 40 + half * 16];
    a.h[1] = *(const v8h*)&Ac[(wm * 16 + l16) * 40 + half * 16 + 8];
    const int kb2 = half * 16;
    F16F b0, b1;
    b0.h[0] = *(const v8h*)&Bc[((wn * 2 + 0) * 16 + l16) * 40 + kb2];
    b0.h[1] = *(const v8h*)&Bc[((wn * 2 + 0) * 16 + l16) * 40 + kb2 + 8];
    b1.h[0] = *(const v8h*)&Bc[((wn * 2 + 1) * 16 + l16) * 40 + kb2];
    b1.h[1] = *(const v8h*)&Bc[((wn * 2 + 1) * 16 + l16) * 40 + kb2 + 8];
    acc0 = wmma_f16(a.v, b0.v, acc0);
    acc1 = wmma_f16(a.v, b1.v, acc1);
  }
  const long long ofs = (long long)b * bof;
  const long long o16 = (long long)b * bo16;
  for (int t = 0; t < 2; ++t) {
    v8f acc = t ? acc1 : acc0;
    const int col = n0 + (wn * 2 + t) * 16 + l16;
    const int row0 = m0 + wm * 16 + half * 8;
    float vals[8];
    #pragma unroll
    for (int r = 0; r < 8; ++r) {
      float v = acc[r];
      int row = row0 + r;
      if (row < M) {
        if (mode == 0) v += p0[row];
        else { v = v * p0[row] + p1[row]; if (mode == 2) v = fmaxf(v, 0.f); }
      }
      vals[r] = v;
    }
    if (Outf) {
      #pragma unroll
      for (int r = 0; r < 8; ++r)
        if (row0 + r < M) Outf[ofs + (long long)(row0 + r) * N + col] = vals[r];
    }
    if (Out16T && row0 + 8 <= M) {            // one b128 store for 8 chained rows
      v8h pk;
      #pragma unroll
      for (int r = 0; r < 8; ++r) pk[r] = (_Float16)vals[r];
      *(v8h*)&Out16T[o16 + (long long)col * M + row0] = pk;
    }
    if (Out16RM) {
      #pragma unroll
      for (int r = 0; r < 8; ++r)
        if (row0 + r < M) Out16RM[o16 + (long long)(row0 + r) * N + col] = (_Float16)vals[r];
    }
  }
}

// ---------------------------------------------------------------------------
// Flash attention. grid=(HW/64, NH, B), block=128 (4 waves, 16 q-rows each).
// S via f32 WMMA 16x16x4 (exact logits); online softmax; P@V via f16 WMMA.
// Output: ct16T[b][i][head*32+d] (transposed f16, ready as o-proj GEMM B input).
__global__ __launch_bounds__(128) void attn_flash(
    const float* __restrict__ q1g, const float* __restrict__ q2g,
    const float* __restrict__ kg, const _Float16* __restrict__ vg,
    _Float16* __restrict__ ct) {
  __shared__ __attribute__((aligned(16))) float Qs[64 * 8];
  __shared__ __attribute__((aligned(16))) float Kc[8 * 33];
  __shared__ __attribute__((aligned(16))) _Float16 VT[32 * 40];
  __shared__ __attribute__((aligned(16))) _Float16 Ps[4 * 16 * 40];
  const int b = blockIdx.z, head = blockIdx.y;
  const int r0 = blockIdx.x * 64;
  const int tid = threadIdx.x;
  const int lane = tid & 31, wave = tid >> 5;
  const int half = lane >> 4, l16 = lane & 15;
  {  // stage concat q (two-phase so loads batch)
    float tq[4];
    #pragma unroll
    for (int j = 0; j < 4; ++j) {
      int i = tid + j * 128;
      int rl = i >> 3, c = i & 7;
      int gi = r0 + rl;
      tq[j] = (c < 4) ? q1g[((long long)b * 16 + head * 4 + c) * HWSZ + gi]
                      : q2g[((long long)b * 16 + head * 4 + (c - 4)) * HWSZ + gi];
    }
    #pragma unroll
    for (int j = 0; j < 4; ++j) {
      int i = tid + j * 128;
      Qs[(i >> 3) * 8 + (i & 7)] = tq[j];
    }
  }
  float rm[8], rl_[8];
  #pragma unroll
  for (int r = 0; r < 8; ++r) { rm[r] = -1e30f; rl_[r] = 0.f; }
  v8f o0 = {}, o1 = {};
  const float sc = 0.1767766952966369f;  // 1/sqrt(32)
  _Float16* Pw = &Ps[wave * 16 * 40];
  const int vd = tid >> 2, vjc = (tid & 3) * 8;   // V staging map: 32 d x 4 chunks
  for (int j0 = 0; j0 < HWSZ; j0 += 32) {
    __syncthreads();
    {  // K^T slice (f32, 8x32) two-phase
      float t0 = kg[((long long)b * 32 + head * 8 + (tid >> 5) + 0) * HWSZ + j0 + (tid & 31)];
      float t1 = kg[((long long)b * 32 + head * 8 + (tid >> 5) + 4) * HWSZ + j0 + (tid & 31)];
      Kc[((tid >> 5) + 0) * 33 + (tid & 31)] = t0;
      Kc[((tid >> 5) + 4) * 33 + (tid & 31)] = t1;
    }
    // V slice [32 d][32 j] f16, contiguous 16B chunks
#if USE_ASYNC_LDS
    async_b128((unsigned)(size_t)&VT[vd * 40 + vjc],
               &vg[((long long)b * 128 + head * 32 + vd) * HWSZ + j0 + vjc]);
    wait_async_n<0>();
#else
    *(v8h*)&VT[vd * 40 + vjc] =
        *(const v8h*)&vg[((long long)b * 128 + head * 32 + vd) * HWSZ + j0 + vjc];
#endif
    __syncthreads();
    v8f s0 = {}, s1 = {};
#if defined(HAVE_WMMA_F32X4)
    #pragma unroll
    for (int koff = 0; koff < 8; koff += 4) {
      v2f a, b0, b1;
      int arow = wave * 16 + l16;
      a.x = Qs[arow * 8 + koff + half * 2];
      a.y = Qs[arow * 8 + koff + half * 2 + 1];
      b0.x = Kc[(koff + half * 2) * 33 + l16];
      b0.y = Kc[(koff + half * 2 + 1) * 33 + l16];
      b1.x = Kc[(koff + half * 2) * 33 + 16 + l16];
      b1.y = Kc[(koff + half * 2 + 1) * 33 + 16 + l16];
      s0 = wmma_f32x4(a, b0, s0);
      s1 = wmma_f32x4(a, b1, s1);
    }
#else
    #pragma unroll
    for (int r = 0; r < 8; ++r) {
      int row = wave * 16 + r + half * 8;
      float a0 = 0.f, a1 = 0.f;
      for (int c = 0; c < 8; ++c) {
        float qv = Qs[row * 8 + c];
        a0 += qv * Kc[c * 33 + l16];
        a1 += qv * Kc[c * 33 + 16 + l16];
      }
      s0[r] = a0; s1[r] = a1;
    }
#endif
    // Online softmax; row stats live in 16-lane halves (C-layout striping).
    #pragma unroll
    for (int r = 0; r < 8; ++r) {
      float v = fmaxf(s0[r], s1[r]) * sc;
      v = fmaxf(v, __shfl_xor(v, 8, 32));
      v = fmaxf(v, __shfl_xor(v, 4, 32));
      v = fmaxf(v, __shfl_xor(v, 2, 32));
      v = fmaxf(v, __shfl_xor(v, 1, 32));
      float mnew = fmaxf(rm[r], v);
      float corr = __expf(rm[r] - mnew);
      rm[r] = mnew;
      float p0v = __expf(s0[r] * sc - mnew);
      float p1v = __expf(s1[r] * sc - mnew);
      float ps = p0v + p1v;
      ps += __shfl_xor(ps, 8, 32);
      ps += __shfl_xor(ps, 4, 32);
      ps += __shfl_xor(ps, 2, 32);
      ps += __shfl_xor(ps, 1, 32);
      rl_[r] = rl_[r] * corr + ps;
      o0[r] *= corr; o1[r] *= corr;
      int prow = r + half * 8;
      Pw[prow * 40 + l16] = (_Float16)p0v;
      Pw[prow * 40 + 16 + l16] = (_Float16)p1v;
    }
    __syncthreads();
    F16F pa, vb0, vb1;
    pa.h[0] = *(const v8h*)&Pw[l16 * 40 + half * 16];
    pa.h[1] = *(const v8h*)&Pw[l16 * 40 + half * 16 + 8];
    const int kb2 = half * 16;
    vb0.h[0] = *(const v8h*)&VT[l16 * 40 + kb2];
    vb0.h[1] = *(const v8h*)&VT[l16 * 40 + kb2 + 8];
    vb1.h[0] = *(const v8h*)&VT[(16 + l16) * 40 + kb2];
    vb1.h[1] = *(const v8h*)&VT[(16 + l16) * 40 + kb2 + 8];
    o0 = wmma_f16(pa.v, vb0.v, o0);
    o1 = wmma_f16(pa.v, vb1.v, o1);
  }
  #pragma unroll
  for (int r = 0; r < 8; ++r) {
    float inv = 1.f / rl_[r];
    int gi = r0 + wave * 16 + r + half * 8;
    int d0 = head * 32 + l16;
    ct[((long long)b * HWSZ + gi) * 128 + d0] = (_Float16)(o0[r] * inv);
    ct[((long long)b * HWSZ + gi) * 128 + d0 + 16] = (_Float16)(o1[r] * inv);
  }
}

// ---------------------------------------------------------------------------
// gamma*o + x residual, then LN over raw-view groups of 128 (= GEMM rows).
__global__ __launch_bounds__(128) void ln_res(
    const float* __restrict__ oproj, const float* __restrict__ xres,
    const float* __restrict__ gamma, const float* __restrict__ lw,
    const float* __restrict__ lb, float* __restrict__ outb) {
  __shared__ float red[4];
  const int b = blockIdx.y, i = blockIdx.x, c = threadIdx.x;
  const int lane = c & 31, wave = c >> 5;
  float o = oproj[((long long)b * 128 + c) * HWSZ + i];
  int ch = i / 18, pos = (i % 18) * 128 + c;
  float t = gamma[0] * o + xres[((long long)b * 128 + ch) * HWSZ + pos];
  float s = t;
  s += __shfl_xor(s, 16, 32); s += __shfl_xor(s, 8, 32); s += __shfl_xor(s, 4, 32);
  s += __shfl_xor(s, 2, 32);  s += __shfl_xor(s, 1, 32);
  if (lane == 0) red[wave] = s;
  __syncthreads();
  float mean = (red[0] + red[1] + red[2] + red[3]) * (1.f / 128.f);
  __syncthreads();
  float d = t - mean;
  float s2 = d * d;
  s2 += __shfl_xor(s2, 16, 32); s2 += __shfl_xor(s2, 8, 32); s2 += __shfl_xor(s2, 4, 32);
  s2 += __shfl_xor(s2, 2, 32);  s2 += __shfl_xor(s2, 1, 32);
  if (lane == 0) red[wave] = s2;
  __syncthreads();
  float var = (red[0] + red[1] + red[2] + red[3]) * (1.f / 128.f);
  float y = d * rsqrtf(var + EPSV) * lw[c] + lb[c];
  outb[(long long)b * 128 * HWSZ + (long long)i * 128 + c] = y;
}

// ---------------------------------------------------------------------------
// im2col (3x3, SAME) of concat(o1,o2) -> f16 transposed [p][ic*9+kp].
__global__ __launch_bounds__(256) void im2col3(
    const float* __restrict__ o1, const float* __restrict__ o2,
    _Float16* __restrict__ im) {
  const int b = blockIdx.y;
  const int p = blockIdx.x;
  const int ic = threadIdx.x;  // 0..255
  const float* row = ((ic < 128) ? o1 : o2) + ((long long)b * 128 + (ic & 127)) * HWSZ;
  const int h = p / 48, w = p % 48;
  _Float16* dst = im + ((long long)b * HWSZ + p) * 2304 + ic * 9;
  #pragma unroll
  for (int kp = 0; kp < 9; ++kp) {
    int hh = h + kp / 3 - 1, ww = w + kp % 3 - 1;
    float v = (hh >= 0 && hh < 48 && ww >= 0 && ww < 48) ? row[hh * 48 + ww] : 0.f;
    dst[kp] = (_Float16)v;
  }
}

// ---------------------------------------------------------------------------
extern "C" void kernel_launch(void* const* d_in, const int* in_sizes, int n_in,
                              void* d_out, int out_size, void* d_ws, size_t ws_size,
                              hipStream_t stream) {
  (void)in_sizes; (void)n_in; (void)out_size; (void)ws_size;
  const float* input1 = (const float*)d_in[0];
  const float* input2 = (const float*)d_in[1];
  const float* ts_w = (const float*)d_in[2];  const float* ts_b = (const float*)d_in[3];
  const float* ts_g = (const float*)d_in[4];  const float* ts_be = (const float*)d_in[5];
  const float* ts_m = (const float*)d_in[6];  const float* ts_v = (const float*)d_in[7];
  const float* tq_w = (const float*)d_in[8];  const float* tq_b = (const float*)d_in[9];
  const float* tq_g = (const float*)d_in[10]; const float* tq_be = (const float*)d_in[11];
  const float* tq_m = (const float*)d_in[12]; const float* tq_v = (const float*)d_in[13];
  const float* q1_w = (const float*)d_in[14]; const float* q1_b = (const float*)d_in[15];
  const float* k1_w = (const float*)d_in[16]; const float* k1_b = (const float*)d_in[17];
  const float* v1_w = (const float*)d_in[18]; const float* v1_b = (const float*)d_in[19];
  const float* q2_w = (const float*)d_in[20]; const float* q2_b = (const float*)d_in[21];
  const float* k2_w = (const float*)d_in[22]; const float* k2_b = (const float*)d_in[23];
  const float* v2_w = (const float*)d_in[24]; const float* v2_b = (const float*)d_in[25];
  const float* gamma1 = (const float*)d_in[26]; const float* gamma2 = (const float*)d_in[27];
  const float* wo1_w = (const float*)d_in[28]; const float* wo1_b = (const float*)d_in[29];
  const float* wo2_w = (const float*)d_in[30]; const float* wo2_b = (const float*)d_in[31];
  const float* ln_w = (const float*)d_in[32];  const float* ln_b = (const float*)d_in[33];
  const float* cat_w = (const float*)d_in[34];
  const float* cat_g = (const float*)d_in[35]; const float* cat_be = (const float*)d_in[36];
  const float* cat_m = (const float*)d_in[37]; const float* cat_v = (const float*)d_in[38];

  const long long S_CH = (long long)CI * HWSZ;  // 294912
  char* cur = (char*)d_ws;
  auto alloc = [&](size_t bytes) {
    char* p = cur; cur += (bytes + 255) & ~(size_t)255; return p;
  };
  float* X1  = (float*)alloc(BB * S_CH * 4);
  float* X2  = (float*)alloc(BB * S_CH * 4);
  float* Q1  = (float*)alloc(BB * 16 * HWSZ * 4);
  float* Q2  = (float*)alloc(BB * 16 * HWSZ * 4);
  float* K1  = (float*)alloc(BB * 32 * HWSZ * 4);
  float* K2  = (float*)alloc(BB * 32 * HWSZ * 4);
  float* OP1 = (float*)alloc(BB * S_CH * 4);
  float* OP2 = (float*)alloc(BB * S_CH * 4);
  float* SC  = (float*)alloc(768 * 4);
  _Float16* IN1T = (_Float16*)alloc(BB * (size_t)HWSZ * 256 * 2);
  _Float16* IN2T = (_Float16*)alloc(BB * (size_t)HWSZ * 256 * 2);
  _Float16* X1T  = (_Float16*)alloc(BB * (size_t)HWSZ * 128 * 2);
  _Float16* X2T  = (_Float16*)alloc(BB * (size_t)HWSZ * 128 * 2);
  _Float16* V1h  = (_Float16*)alloc(BB * S_CH * 2);
  _Float16* V2h  = (_Float16*)alloc(BB * S_CH * 2);
  _Float16* CT1T = (_Float16*)alloc(BB * (size_t)HWSZ * 128 * 2);
  _Float16* CT2T = (_Float16*)alloc(BB * (size_t)HWSZ * 128 * 2);
  _Float16* IMT  = (_Float16*)alloc(BB * (size_t)HWSZ * 2304 * 2);
  _Float16* ts16 = (_Float16*)alloc(128 * 256 * 2);
  _Float16* tq16 = (_Float16*)alloc(128 * 256 * 2);
  _Float16* q116 = (_Float16*)alloc(16 * 128 * 2);
  _Float16* k116 = (_Float16*)alloc(32 * 128 * 2);
  _Float16* v116 = (_Float16*)alloc(128 * 128 * 2);
  _Float16* q216 = (_Float16*)alloc(16 * 128 * 2);
  _Float16* k216 = (_Float16*)alloc(32 * 128 * 2);
  _Float16* v216 = (_Float16*)alloc(128 * 128 * 2);
  _Float16* wo116 = (_Float16*)alloc(128 * 128 * 2);
  _Float16* wo216 = (_Float16*)alloc(128 * 128 * 2);
  _Float16* cat16 = (_Float16*)alloc((size_t)128 * 2304 * 2);

  float* yout = (float*)d_out;
  float* o1out = yout + 2 * S_CH;
  float* o2out = o1out + 2 * S_CH;

  bn_prep<<<1, 128, 0, stream>>>(ts_g, ts_be, ts_m, ts_v, ts_b, SC, SC + 128, 128);
  bn_prep<<<1, 128, 0, stream>>>(tq_g, tq_be, tq_m, tq_v, tq_b, SC + 256, SC + 384, 128);
  bn_prep<<<1, 128, 0, stream>>>(cat_g, cat_be, cat_m, cat_v, nullptr, SC + 512, SC + 640, 128);

  auto cvt = [&](const float* src, _Float16* dst, int n) {
    cvt_f16<<<(n + 255) / 256, 256, 0, stream>>>(src, dst, n);
  };
  cvt(ts_w, ts16, 128 * 256);  cvt(tq_w, tq16, 128 * 256);
  cvt(q1_w, q116, 16 * 128);   cvt(k1_w, k116, 32 * 128);  cvt(v1_w, v116, 128 * 128);
  cvt(q2_w, q216, 16 * 128);   cvt(k2_w, k216, 32 * 128);  cvt(v2_w, v216, 128 * 128);
  cvt(wo1_w, wo116, 128 * 128); cvt(wo2_w, wo216, 128 * 128);
  cvt(cat_w, cat16, 128 * 2304);

  transpose_cvt<<<dim3(72, 8, BB), 256, 0, stream>>>(input1, IN1T, 256, HWSZ,
                                                     (long long)CIN * HWSZ, (long long)HWSZ * 256);
  transpose_cvt<<<dim3(72, 8, BB), 256, 0, stream>>>(input2, IN2T, 256, HWSZ,
                                                     (long long)CIN * HWSZ, (long long)HWSZ * 256);

  dim3 blk(256);
  // stems: x = BN(conv1x1(input)); f32 for residual + f16T for chaining
  gemm_wf16<<<dim3(36, 2, BB), blk, 0, stream>>>(ts16, IN1T, X1, X1T, nullptr,
      SC, SC + 128, 128, 256, HWSZ, (long long)HWSZ * 256, S_CH, (long long)HWSZ * 128, 1);
  gemm_wf16<<<dim3(36, 2, BB), blk, 0, stream>>>(tq16, IN2T, X2, X2T, nullptr,
      SC + 256, SC + 384, 128, 256, HWSZ, (long long)HWSZ * 256, S_CH, (long long)HWSZ * 128, 1);
  // projections
  gemm_wf16<<<dim3(36, 1, BB), blk, 0, stream>>>(q116, X1T, Q1, nullptr, nullptr,
      q1_b, nullptr, 16, 128, HWSZ, (long long)HWSZ * 128, 16LL * HWSZ, 0, 0);
  gemm_wf16<<<dim3(36, 1, BB), blk, 0, stream>>>(k116, X1T, K1, nullptr, nullptr,
      k1_b, nullptr, 32, 128, HWSZ, (long long)HWSZ * 128, 32LL * HWSZ, 0, 0);
  gemm_wf16<<<dim3(36, 2, BB), blk, 0, stream>>>(v116, X1T, nullptr, nullptr, V1h,
      v1_b, nullptr, 128, 128, HWSZ, (long long)HWSZ * 128, 0, S_CH, 0);
  gemm_wf16<<<dim3(36, 1, BB), blk, 0, stream>>>(q216, X2T, Q2, nullptr, nullptr,
      q2_b, nullptr, 16, 128, HWSZ, (long long)HWSZ * 128, 16LL * HWSZ, 0, 0);
  gemm_wf16<<<dim3(36, 1, BB), blk, 0, stream>>>(k216, X2T, K2, nullptr, nullptr,
      k2_b, nullptr, 32, 128, HWSZ, (long long)HWSZ * 128, 32LL * HWSZ, 0, 0);
  gemm_wf16<<<dim3(36, 2, BB), blk, 0, stream>>>(v216, X2T, nullptr, nullptr, V2h,
      v2_b, nullptr, 128, 128, HWSZ, (long long)HWSZ * 128, 0, S_CH, 0);
  // flash attention
  attn_flash<<<dim3(36, NHEAD, BB), dim3(128), 0, stream>>>(Q1, Q2, K1, V1h, CT1T);
  attn_flash<<<dim3(36, NHEAD, BB), dim3(128), 0, stream>>>(Q1, Q2, K2, V2h, CT2T);
  // output projections
  gemm_wf16<<<dim3(36, 2, BB), blk, 0, stream>>>(wo116, CT1T, OP1, nullptr, nullptr,
      wo1_b, nullptr, 128, 128, HWSZ, (long long)HWSZ * 128, S_CH, 0, 0);
  gemm_wf16<<<dim3(36, 2, BB), blk, 0, stream>>>(wo216, CT2T, OP2, nullptr, nullptr,
      wo2_b, nullptr, 128, 128, HWSZ, (long long)HWSZ * 128, S_CH, 0, 0);
  // residual + LN (write o1/o2 outputs)
  ln_res<<<dim3(HWSZ, BB), dim3(128), 0, stream>>>(OP1, X1, gamma1, ln_w, ln_b, o1out);
  ln_res<<<dim3(HWSZ, BB), dim3(128), 0, stream>>>(OP2, X2, gamma2, ln_w, ln_b, o2out);
  // im2col + conv-as-GEMM + BN + ReLU
  im2col3<<<dim3(HWSZ, BB), 256, 0, stream>>>(o1out, o2out, IMT);
  gemm_wf16<<<dim3(36, 2, BB), blk, 0, stream>>>(cat16, IMT, yout, nullptr, nullptr,
      SC + 512, SC + 640, 128, 2304, HWSZ, (long long)HWSZ * 2304, S_CH, 0, 2);
}
